// QuaternionicAttentionRetriever_32959579029634
// MI455X (gfx1250) — matrix operand order repaired and checked by reading, not compiled, not run on hardware
//
#include <hip/hip_runtime.h>
#include <math.h>

// ---------------- problem constants ----------------
#define B_DIM   8
#define S_DIM   2048
#define D_DIM   1024
#define LOG2S   11
#define PI_F    3.14159265358979323846f

typedef __attribute__((ext_vector_type(16))) __bf16 v16bf;
typedef __attribute__((ext_vector_type(8)))  __bf16 v8bf;
typedef __attribute__((ext_vector_type(4)))  __bf16 v4bf;
typedef __attribute__((ext_vector_type(8)))  float  v8f;

// ---------------- complex helpers ----------------
__device__ __forceinline__ float2 cmul(float2 a, float2 b) {
    return make_float2(a.x * b.x - a.y * b.y, a.x * b.y + a.y * b.x);
}
__device__ __forceinline__ float2 cadd(float2 a, float2 b) {
    return make_float2(a.x + b.x, a.y + b.y);
}
__device__ __forceinline__ float2 csub(float2 a, float2 b) {
    return make_float2(a.x - b.x, a.y - b.y);
}

// A*B ~= Ahi*Bhi + Ahi*Blo + Alo*Bhi  (near-fp32 accuracy at bf16 rate)
__device__ __forceinline__ v8f mma3(v16bf ah, v16bf al, v16bf bh, v16bf bl, v8f c) {
    c = __builtin_amdgcn_wmma_f32_16x16x32_bf16(false, ah, false, bh, (short)0, c, false, false);
    c = __builtin_amdgcn_wmma_f32_16x16x32_bf16(false, ah, false, bl, (short)0, c, false, false);
    c = __builtin_amdgcn_wmma_f32_16x16x32_bf16(false, al, false, bh, (short)0, c, false, false);
    return c;
}

// ================================================================
// Kernel 0a: transpose 1024x1024 weights AND split fp32 -> bf16 hi/lo
// planes so GEMM B-fragments are contiguous bf16 runs per lane.
// ================================================================
__global__ __launch_bounds__(256)
void transpose_split1024(const float* __restrict__ src,
                         __bf16* __restrict__ hi, __bf16* __restrict__ lo) {
    __shared__ float tile[32][33];
    int x  = blockIdx.x * 32 + threadIdx.x;
    int y0 = blockIdx.y * 32 + threadIdx.y;
    #pragma unroll
    for (int j = 0; j < 32; j += 8)
        tile[threadIdx.y + j][threadIdx.x] = src[(size_t)(y0 + j) * D_DIM + x];
    __syncthreads();
    int x2 = blockIdx.y * 32 + threadIdx.x;
    int y2 = blockIdx.x * 32 + threadIdx.y;
    #pragma unroll
    for (int j = 0; j < 32; j += 8) {
        float v = tile[threadIdx.x][threadIdx.y + j];
        __bf16 h = (__bf16)v;
        size_t o = (size_t)(y2 + j) * D_DIM + x2;
        hi[o] = h;
        lo[o] = (__bf16)(v - (float)h);
    }
}

// ================================================================
// Kernel 0b: streaming fp32 -> bf16 hi/lo split for activations.
// Each thread handles 4 elements (float4 in, 2x v4bf out).
// ================================================================
__global__ __launch_bounds__(256)
void split_bf16(const float* __restrict__ src,
                __bf16* __restrict__ hi, __bf16* __restrict__ lo, int n4) {
    int idx = blockIdx.x * 256 + threadIdx.x;
    if (idx >= n4) return;
    float4 x = reinterpret_cast<const float4*>(src)[idx];
    float f[4] = {x.x, x.y, x.z, x.w};
    v4bf h, l;
    #pragma unroll
    for (int i = 0; i < 4; ++i) {
        __bf16 hh = (__bf16)f[i];
        h[i] = hh;
        l[i] = (__bf16)(f[i] - (float)hh);
    }
    reinterpret_cast<v4bf*>(hi)[idx] = h;
    reinterpret_cast<v4bf*>(lo)[idx] = l;
}

// ---------------- fragment loaders (pre-split bf16) ----------------
// A fragment: ISA layout lane<16: K {0..7,16..23}, lane>=16: K {8..15,24..31}
__device__ __forceinline__ v16bf load_afrag(const __bf16* __restrict__ base) {
    v8bf a = *reinterpret_cast<const v8bf*>(base);
    v8bf b = *reinterpret_cast<const v8bf*>(base + 16);
    v16bf r;
    #pragma unroll
    for (int i = 0; i < 8; ++i) { r[i] = a[i]; r[8 + i] = b[i]; }
    return r;
}
// B fragment: lane<16: K {0..7,16..23}, lane>=16: K {8..15,24..31}
__device__ __forceinline__ v16bf load_bfrag(const __bf16* __restrict__ base) {
    return load_afrag(base);
}

// ================================================================
// Kernel 1: GEMM  C[M,N] = A[M,K] * W[K,N] + bias   (M=16384, N=K=1024)
// Pre-split bf16 operands -> inner loop is pure loads + 24 WMMAs per
// 32-K chunk. Each wave computes a 32x64 block (8 accumulators).
// ================================================================
__global__ __launch_bounds__(256)
void gemm_wmma_bf16x3(const __bf16* __restrict__ Ahi, const __bf16* __restrict__ Alo,
                      const __bf16* __restrict__ Bhi, const __bf16* __restrict__ Blo,
                      const float* __restrict__ bias, float* __restrict__ C) {
    const int lane = threadIdx.x & 31;
    const int wave = threadIdx.x >> 5;
    const int M0   = blockIdx.x * 32;                 // 32-row band
    const int N0   = blockIdx.y * 512 + wave * 64;    // 64-col block per wave

    const int l15  = lane & 15;
    const int aK   = (lane < 16) ? 0 : 8;
    const int bK   = (lane < 16) ? 0 : 8;

    const size_t aOff0 = (size_t)(M0 + l15) * D_DIM + aK;
    const size_t aOff1 = aOff0 + (size_t)16 * D_DIM;
    size_t bOff[4];
    #pragma unroll
    for (int n = 0; n < 4; ++n)
        bOff[n] = (size_t)(N0 + n * 16 + l15) * D_DIM + bK;

    v8f acc[2][4] = {};

    for (int kb = 0; kb < D_DIM; kb += 32) {
        if (kb + 32 < D_DIM) {
            __builtin_prefetch(Ahi + aOff0 + kb + 32, 0, 1);
            __builtin_prefetch(Bhi + bOff[0] + kb + 32, 0, 1);
        }

        v16bf ah[2], al[2];
        ah[0] = load_afrag(Ahi + aOff0 + kb);
        al[0] = load_afrag(Alo + aOff0 + kb);
        ah[1] = load_afrag(Ahi + aOff1 + kb);
        al[1] = load_afrag(Alo + aOff1 + kb);

        v16bf bh[4], bl[4];
        #pragma unroll
        for (int n = 0; n < 4; ++n) {
            bh[n] = load_bfrag(Bhi + bOff[n] + kb);
            bl[n] = load_bfrag(Blo + bOff[n] + kb);
        }

        #pragma unroll
        for (int m = 0; m < 2; ++m)
            #pragma unroll
            for (int n = 0; n < 4; ++n)
                acc[m][n] = mma3(ah[m], al[m], bh[n], bl[n], acc[m][n]);
    }

    // C/D layout: VGPR r holds row (r + (lane>=16 ? 8:0)), col = lane&15
    #pragma unroll
    for (int m = 0; m < 2; ++m) {
        const int mBase = M0 + m * 16 + ((lane >= 16) ? 8 : 0);
        #pragma unroll
        for (int n = 0; n < 4; ++n) {
            const int nCol = N0 + n * 16 + l15;
            const float bv = bias[nCol];
            #pragma unroll
            for (int r = 0; r < 8; ++r)
                C[(size_t)(mBase + r) * D_DIM + nCol] = acc[m][n][r] + bv;
        }
    }
}

// ================================================================
// Kernel 2: forward 2048-pt FFT along seq per (b,d) column + spectral
// filter; writes complex spectrum in [b][d][s] layout (s contiguous).
// ================================================================
__global__ __launch_bounds__(256)
void fft_fwd_filter(const float* __restrict__ src, float2* __restrict__ dst) {
    const int d   = blockIdx.x;
    const int b   = blockIdx.y;
    const int tid = threadIdx.x;
    __shared__ float2 buf[S_DIM];   // 16 KB

    for (int i = tid; i < S_DIM; i += 256) {
        float x = src[((size_t)(b * S_DIM + i)) * D_DIM + d];
        int r = (int)(__brev((unsigned)i) >> (32 - LOG2S));
        buf[r] = make_float2(x, 0.0f);
    }
    __syncthreads();

    for (int len = 1; len < S_DIM; len <<= 1) {
        const float ang = -PI_F / (float)len;
        for (int j = tid; j < S_DIM / 2; j += 256) {
            int pos = j & (len - 1);
            int i1  = ((j & ~(len - 1)) << 1) + pos;
            int i2  = i1 + len;
            float s, c;
            __sincosf(ang * (float)pos, &s, &c);
            float2 u = buf[i1], v = buf[i2];
            float2 t = make_float2(v.x * c - v.y * s, v.x * s + v.y * c);
            buf[i1] = make_float2(u.x + t.x, u.y + t.y);
            buf[i2] = make_float2(u.x - t.x, u.y - t.y);
        }
        __syncthreads();
    }

    float2* col = dst + ((size_t)(b * D_DIM + d)) * S_DIM;
    for (int i = tid; i < S_DIM; i += 256) {
        float ph = 1.5f * atanf(logf((float)i + 1e-10f));
        float fs, fc;
        __sincosf(ph, &fs, &fc);
        col[i] = cmul(buf[i], make_float2(fc, fs));
    }
}

// ================================================================
// Kernel 3: per (b, e): gather 4 quaternion component spectra of
// Q,K,V, Hamilton-product chain in complex, 4 fused inverse FFTs
// in LDS (64 KB of the 320 KB WGP pool), write real part.
// ================================================================
__global__ __launch_bounds__(256)
void quat_ifft(const float2* __restrict__ Qf, const float2* __restrict__ Kf,
               const float2* __restrict__ Vf, float* __restrict__ out) {
    const int e   = blockIdx.x;   // 0..255
    const int b   = blockIdx.y;   // 0..7
    const int tid = threadIdx.x;
    __shared__ float2 sbuf[4 * S_DIM];   // 64 KB
    float2* P0 = sbuf;
    float2* P1 = sbuf + S_DIM;
    float2* P2 = sbuf + 2 * S_DIM;
    float2* P3 = sbuf + 3 * S_DIM;

    for (int i = tid; i < S_DIM; i += 256) {
        float2 q[4], k[4], v[4];
        #pragma unroll
        for (int c = 0; c < 4; ++c) {
            size_t off = ((size_t)(b * D_DIM + (c * 256 + e))) * S_DIM + i;
            q[c] = Qf[off]; k[c] = Kf[off]; v[c] = Vf[off];
        }
        float2 t[4], p[4];
        t[0] = csub(csub(csub(cmul(q[0], k[0]), cmul(q[1], k[1])), cmul(q[2], k[2])), cmul(q[3], k[3]));
        t[1] = csub(cadd(cadd(cmul(q[0], k[1]), cmul(q[1], k[0])), cmul(q[2], k[3])), cmul(q[3], k[2]));
        t[2] = cadd(cadd(csub(cmul(q[0], k[2]), cmul(q[1], k[3])), cmul(q[2], k[0])), cmul(q[3], k[1]));
        t[3] = cadd(csub(cadd(cmul(q[0], k[3]), cmul(q[1], k[2])), cmul(q[2], k[1])), cmul(q[3], k[0]));
        p[0] = csub(csub(csub(cmul(t[0], v[0]), cmul(t[1], v[1])), cmul(t[2], v[2])), cmul(t[3], v[3]));
        p[1] = csub(cadd(cadd(cmul(t[0], v[1]), cmul(t[1], v[0])), cmul(t[2], v[3])), cmul(t[3], v[2]));
        p[2] = cadd(cadd(csub(cmul(t[0], v[2]), cmul(t[1], v[3])), cmul(t[2], v[0])), cmul(t[3], v[1]));
        p[3] = cadd(csub(cadd(cmul(t[0], v[3]), cmul(t[1], v[2])), cmul(t[2], v[1])), cmul(t[3], v[0]));

        int r = (int)(__brev((unsigned)i) >> (32 - LOG2S));
        P0[r] = p[0]; P1[r] = p[1]; P2[r] = p[2]; P3[r] = p[3];
    }
    __syncthreads();

    for (int len = 1; len < S_DIM; len <<= 1) {
        const float ang = PI_F / (float)len;
        for (int j = tid; j < S_DIM / 2; j += 256) {
            int pos = j & (len - 1);
            int i1  = ((j & ~(len - 1)) << 1) + pos;
            int i2  = i1 + len;
            float s, c;
            __sincosf(ang * (float)pos, &s, &c);
            float2* bufs[4] = {P0, P1, P2, P3};
            #pragma unroll
            for (int q4 = 0; q4 < 4; ++q4) {
                float2 u = bufs[q4][i1], v = bufs[q4][i2];
                float2 t = make_float2(v.x * c - v.y * s, v.x * s + v.y * c);
                bufs[q4][i1] = make_float2(u.x + t.x, u.y + t.y);
                bufs[q4][i2] = make_float2(u.x - t.x, u.y - t.y);
            }
        }
        __syncthreads();
    }

    const float inv = 1.0f / (float)S_DIM;
    for (int i = tid; i < S_DIM; i += 256) {
        size_t rowBase = ((size_t)(b * S_DIM + i)) * D_DIM;
        out[rowBase + (0 * 256 + e)] = P0[i].x * inv;
        out[rowBase + (1 * 256 + e)] = P1[i].x * inv;
        out[rowBase + (2 * 256 + e)] = P2[i].x * inv;
        out[rowBase + (3 * 256 + e)] = P3[i].x * inv;
    }
}

// ================================================================
// Host-side orchestration
// ================================================================
extern "C" void kernel_launch(void* const* d_in, const int* in_sizes, int n_in,
                              void* d_out, int out_size, void* d_ws, size_t ws_size,
                              hipStream_t stream) {
    (void)in_sizes; (void)n_in; (void)out_size; (void)ws_size;

    const float* query = (const float*)d_in[0];
    const float* mem   = (const float*)d_in[1];
    const float* wq    = (const float*)d_in[2];
    const float* bq    = (const float*)d_in[3];
    const float* wk    = (const float*)d_in[4];
    const float* bk    = (const float*)d_in[5];
    const float* wv    = (const float*)d_in[6];
    const float* bv    = (const float*)d_in[7];
    float* out = (float*)d_out;

    // workspace carve-up
    const size_t WSZ = (size_t)D_DIM * D_DIM;           // 1M  elems : weight plane
    const size_t MSZ = (size_t)B_DIM * S_DIM * D_DIM;   // 16M elems : activation matrix
    char* wsb = (char*)d_ws;
    size_t off = 0;
    auto carve = [&](size_t bytes) { char* p = wsb + off; off += bytes; return p; };

    __bf16* whQ = (__bf16*)carve(WSZ * 2);
    __bf16* wlQ = (__bf16*)carve(WSZ * 2);
    __bf16* whK = (__bf16*)carve(WSZ * 2);
    __bf16* wlK = (__bf16*)carve(WSZ * 2);
    __bf16* whV = (__bf16*)carve(WSZ * 2);
    __bf16* wlV = (__bf16*)carve(WSZ * 2);
    __bf16* qh  = (__bf16*)carve(MSZ * 2);   // query hi/lo
    __bf16* ql  = (__bf16*)carve(MSZ * 2);
    __bf16* mh  = (__bf16*)carve(MSZ * 2);   // memory hi/lo
    __bf16* ml  = (__bf16*)carve(MSZ * 2);
    float*  Q   = (float*)carve(MSZ * 4);
    float*  K   = (float*)carve(MSZ * 4);
    float*  V   = (float*)carve(MSZ * 4);
    float2* Qf  = (float2*)carve(MSZ * 8);
    float2* Kf  = (float2*)carve(MSZ * 8);
    float2* Vf  = (float2*)carve(MSZ * 8);

    // 0) operand prep: weight transpose+split, activation split
    {
        dim3 g(D_DIM / 32, D_DIM / 32), blk(32, 8);
        transpose_split1024<<<g, blk, 0, stream>>>(wq, whQ, wlQ);
        transpose_split1024<<<g, blk, 0, stream>>>(wk, whK, wlK);
        transpose_split1024<<<g, blk, 0, stream>>>(wv, whV, wlV);
        int n4 = (int)(MSZ / 4);
        split_bf16<<<(n4 + 255) / 256, 256, 0, stream>>>(query, qh, ql, n4);
        split_bf16<<<(n4 + 255) / 256, 256, 0, stream>>>(mem,   mh, ml, n4);
    }

    // 1) projections via WMMA (32x64 register-blocked waves)
    {
        dim3 g((B_DIM * S_DIM) / 32, D_DIM / 512);
        gemm_wmma_bf16x3<<<g, 256, 0, stream>>>(qh, ql, whQ, wlQ, bq, Q);
        gemm_wmma_bf16x3<<<g, 256, 0, stream>>>(mh, ml, whK, wlK, bk, K);
        gemm_wmma_bf16x3<<<g, 256, 0, stream>>>(mh, ml, whV, wlV, bv, V);
    }

    // 2) forward FFT + spectral filter
    {
        dim3 g(D_DIM, B_DIM);
        fft_fwd_filter<<<g, 256, 0, stream>>>(Q, Qf);
        fft_fwd_filter<<<g, 256, 0, stream>>>(K, Kf);
        fft_fwd_filter<<<g, 256, 0, stream>>>(V, Vf);
    }

    // 3) quaternion Hamilton chain + fused inverse FFTs + real part
    {
        dim3 g(256, B_DIM);
        quat_ifft<<<g, 256, 0, stream>>>(Qf, Kf, Vf, out);
    }
}
